// FractalRecurrentALU_42142219108457
// MI455X (gfx1250) — matrix operand order repaired and checked by reading, not compiled, not run on hardware
//
#include <hip/hip_runtime.h>
#include <hip/hip_bf16.h>
#include <math.h>

// ---------------- problem constants (match reference) ----------------
#define BATCH 1024
#define SEQ   64
#define DIM   512
#define NH    8
#define NL    4
#define FF    2048
#define DHD   64
#define TD    1536          // 3*DIM
#define EPSV  1e-5f
#define SCALEV 0.125f       // 1/sqrt(64)

typedef __bf16 bf16_t;
typedef __bf16 v16bf __attribute__((ext_vector_type(16)));
typedef __bf16 v8bf  __attribute__((ext_vector_type(8)));
typedef float  v8f   __attribute__((ext_vector_type(8)));
typedef int    v4i   __attribute__((ext_vector_type(4)));

typedef __attribute__((address_space(1))) v4i* gv4i_p;   // global v4i*
typedef __attribute__((address_space(3))) v4i* lv4i_p;   // LDS v4i*

// -------- CDNA5 async copy (global -> LDS), builtin or inline asm -----
#if defined(__has_builtin)
#if __has_builtin(__builtin_amdgcn_global_load_async_to_lds_b128)
#define HAVE_ASYNC_BUILTIN 1
#endif
#if __has_builtin(__builtin_amdgcn_s_wait_asynccnt)
#define HAVE_WAIT_BUILTIN 1
#endif
#endif

__device__ __forceinline__ void async_cp_b128(const bf16_t* g, bf16_t* l)
{
#if defined(HAVE_ASYNC_BUILTIN)
  // prototype (from clang): (v4i addrspace(1)*, v4i addrspace(3)*, Ii, Ii)
  __builtin_amdgcn_global_load_async_to_lds_b128(
      (gv4i_p)(v4i*)(void*)const_cast<bf16_t*>(g),
      (lv4i_p)(v4i*)(void*)l, 0, 0);
#else
  // VDST = LDS byte address, VADDR = 64-bit global address  (ISA 15.18.3)
  unsigned loff = (unsigned)(unsigned long long)
      (__attribute__((address_space(3))) void*)l;
  unsigned long long ga = (unsigned long long)(const void*)g;
  asm volatile("global_load_async_to_lds_b128 %0, %1, off"
               :: "v"(loff), "v"(ga) : "memory");
#endif
}

__device__ __forceinline__ void wait_async_le4()
{
#if defined(HAVE_WAIT_BUILTIN)
  __builtin_amdgcn_s_wait_asynccnt(4);
#else
  asm volatile("s_wait_asynccnt 0x4" ::: "memory");
#endif
}
__device__ __forceinline__ void wait_async_0()
{
#if defined(HAVE_WAIT_BUILTIN)
  __builtin_amdgcn_s_wait_asynccnt(0);
#else
  asm volatile("s_wait_asynccnt 0x0" ::: "memory");
#endif
}

// =====================================================================
// WMMA GEMM:  out[M,N] = A[M,K](bf16) * W[N,K]^T(bf16) + bias
//             (+ residual, optional ReLU, f32 or bf16 output)
//
// Block = 256 threads (8 waves) -> 64M x 256N tile.
// Wave  = 32M x 64N register tile: 2x4 f32 accumulators, 8 WMMA / k-step.
// A slab (64 x 128 K, shared by 4 waves) double-buffered in LDS via
// global_load_async_to_lds_b128 (+ s_wait_asynccnt); B read directly
// (contiguous b128 per lane, L2-resident) with global_prefetch.
//
// Fragment layout used here (wave32, contiguous k for A and B):
//   A 16x32 bf16 : lane(0-15)=row M, holds K {16*sel+0..15} (contig 32B)
//   B 32x16 bf16 : lane holds row N of W, K {16*sel+0..15} (contig 32B)
//   C 16x16 f32  : VGPR r -> M = r + 8*sel, N = lane&15
// =====================================================================
#define KC    128                 // K-chunk staged in LDS
#define KPAD  (KC + 8)            // +16B row pad: 16 lanes hit disjoint banks

__global__ __launch_bounds__(256)
void gemm_bf16_wmma(const bf16_t* __restrict__ A, int lda,
                    const bf16_t* __restrict__ Wt, int ldw,
                    const float* __restrict__ bias,
                    const float* __restrict__ residual,
                    float* __restrict__ outF,
                    bf16_t* __restrict__ outB,
                    int M, int N, int K, int relu)
{
  __shared__ bf16_t As[2][64][KPAD];

  const int tid  = threadIdx.x;
  const int wave = tid >> 5;
  const int lane = tid & 31;
  const int sel  = lane >> 4;
  const int lr   = lane & 15;

  const int wm = wave & 1;        // 2 wave rows  (32 M each)
  const int wn = wave >> 1;       // 4 wave cols  (64 N each)
  const int mblk  = blockIdx.y * 64;
  const int nbase = blockIdx.x * 256 + wn * 64;
  (void)M;

  // A-slab async stage: 64 rows x KC cols = 1024 x 16B units, 4 per thread
  const bf16_t* __restrict__ Ablk = A + (size_t)mblk * lda;
  auto stage = [&](int buf, int kc) {
#pragma unroll
    for (int it = 0; it < 4; ++it) {
      const int u   = it * 256 + tid;      // 16-byte unit index
      const int row = u >> 4;              // KC/8 = 16 units per row
      const int c8  = (u & 15) * 8;
      async_cp_b128(Ablk + (size_t)row * lda + kc + c8, &As[buf][row][c8]);
    }
  };

  // B row pointers (4 N-tiles per wave)
  const bf16_t* __restrict__ brow[4];
#pragma unroll
  for (int jn = 0; jn < 4; ++jn)
    brow[jn] = Wt + (size_t)(nbase + 16 * jn + lr) * ldw;

  v8f acc[2][4];
#pragma unroll
  for (int mt = 0; mt < 2; ++mt)
#pragma unroll
    for (int jn = 0; jn < 4; ++jn)
      acc[mt][jn] = (v8f){0.f,0.f,0.f,0.f,0.f,0.f,0.f,0.f};

  union AF { v16bf v; v8bf h[2]; };

  stage(0, 0);
  int buf = 0;
  for (int kc = 0; kc < K; kc += KC) {
    const bool has_next = (kc + KC) < K;
    if (has_next) stage(buf ^ 1, kc + KC);       // prefetch next chunk
    if (has_next) wait_async_le4(); else wait_async_0();
    __syncthreads();                             // A slab (buf) visible

#pragma unroll
    for (int j = 0; j < KC / 32; ++j) {          // 4 k-steps of 32
      const int kk = 32 * j;
      AF af[2];
#pragma unroll
      for (int mt = 0; mt < 2; ++mt) {
        const int row = wm * 32 + mt * 16 + lr;
        af[mt].h[0] = *(const v8bf*)&As[buf][row][kk + 16 * sel];
        af[mt].h[1] = *(const v8bf*)&As[buf][row][kk + 16 * sel + 8];
      }
      v16bf bf[4];
#pragma unroll
      for (int jn = 0; jn < 4; ++jn) {
        bf[jn] = *(const v16bf*)(brow[jn] + kc + kk + 16 * sel);
        __builtin_prefetch(brow[jn] + kc + kk + 64, 0, 3);
      }
#pragma unroll
      for (int mt = 0; mt < 2; ++mt)
#pragma unroll
        for (int jn = 0; jn < 4; ++jn)
          acc[mt][jn] = __builtin_amdgcn_wmma_f32_16x16x32_bf16(
              false, af[mt].v, false, bf[jn], (short)0, acc[mt][jn],
              false, false);
    }
    __syncthreads();                             // done reading buf
    buf ^= 1;
  }

  // ----------------------------- epilogue -----------------------------
#pragma unroll
  for (int mt = 0; mt < 2; ++mt) {
#pragma unroll
    for (int jn = 0; jn < 4; ++jn) {
      const int   n  = nbase + 16 * jn + lr;
      const float bv = bias ? bias[n] : 0.f;
#pragma unroll
      for (int r = 0; r < 8; ++r) {
        const int m = mblk + wm * 32 + mt * 16 + r + 8 * sel;
        float v = acc[mt][jn][r] + bv;
        if (residual) v += residual[(size_t)m * N + n];
        if (relu && v < 0.f) v = 0.f;
        if (outB) outB[(size_t)m * N + n] = (bf16_t)v;
        else      outF[(size_t)m * N + n] = v;
      }
    }
  }
}

// ---------------- LayerNorm (one block = one row of 512) -> bf16 ------
__global__ __launch_bounds__(256)
void ln_to_bf16(const float* __restrict__ x, const float* __restrict__ s,
                const float* __restrict__ b, bf16_t* __restrict__ y)
{
  __shared__ float red[256];
  const int   row = blockIdx.x;
  const int   t   = threadIdx.x;
  const float* xr = x + (size_t)row * DIM;
  const float v0 = xr[t], v1 = xr[t + 256];

  red[t] = v0 + v1;
  __syncthreads();
  for (int off = 128; off > 0; off >>= 1) {
    if (t < off) red[t] += red[t + off];
    __syncthreads();
  }
  const float mean = red[0] * (1.f / DIM);
  __syncthreads();

  red[t] = v0 * v0 + v1 * v1;
  __syncthreads();
  for (int off = 128; off > 0; off >>= 1) {
    if (t < off) red[t] += red[t + off];
    __syncthreads();
  }
  const float var = red[0] * (1.f / DIM) - mean * mean;
  const float rs  = rsqrtf(var + EPSV);

  bf16_t* yr = y + (size_t)row * DIM;
  yr[t]       = (bf16_t)((v0 - mean) * rs * s[t]       + b[t]);
  yr[t + 256] = (bf16_t)((v1 - mean) * rs * s[t + 256] + b[t + 256]);
}

// ------------- tiny 3x3 attention, one thread per (batch, head) -------
__global__ void attn_small(const float* __restrict__ qkv,
                           float* __restrict__ attout)
{
  const int idx = blockIdx.x * blockDim.x + threadIdx.x;
  if (idx >= BATCH * NH) return;
  const int b = idx / NH, h = idx % NH;
  const float* base = qkv + (size_t)b * 3 * TD + h * DHD;

  float sc[3][3];
  for (int i = 0; i < 3; ++i)
    for (int j = 0; j < 3; ++j) {
      const float* q = base + (size_t)i * TD;          // q block
      const float* k = base + (size_t)j * TD + DIM;    // k block
      float d = 0.f;
      for (int e = 0; e < DHD; ++e) d += q[e] * k[e];
      sc[i][j] = d * SCALEV;
    }
  for (int i = 0; i < 3; ++i) {
    float mx = fmaxf(sc[i][0], fmaxf(sc[i][1], sc[i][2]));
    float e0 = __expf(sc[i][0] - mx), e1 = __expf(sc[i][1] - mx),
          e2 = __expf(sc[i][2] - mx);
    float inv = 1.f / (e0 + e1 + e2);
    sc[i][0] = e0 * inv; sc[i][1] = e1 * inv; sc[i][2] = e2 * inv;
  }
  for (int i = 0; i < 3; ++i) {
    float* o = attout + ((size_t)b * 3 + i) * DIM + h * DHD;
    const float* v0 = base + 2 * DIM;
    for (int e = 0; e < DHD; ++e)
      o[e] = sc[i][0] * v0[e] + sc[i][1] * v0[TD + e] + sc[i][2] * v0[2 * TD + e];
  }
}

// ---------------- x = stack(h, emb[a], emb[b]) ------------------------
__global__ void build_x(const float* __restrict__ h, const int* __restrict__ a_seq,
                        const int* __restrict__ b_seq, const float* __restrict__ emb,
                        float* __restrict__ x, int t)
{
  const int idx = blockIdx.x * blockDim.x + threadIdx.x;
  if (idx >= BATCH * DIM) return;
  const int b = idx / DIM, d = idx % DIM;
  const int ia = a_seq[b * SEQ + t], ib = b_seq[b * SEQ + t];
  x[((size_t)b * 3 + 0) * DIM + d] = h[idx];
  x[((size_t)b * 3 + 1) * DIM + d] = emb[(size_t)ia * DIM + d];
  x[((size_t)b * 3 + 2) * DIM + d] = emb[(size_t)ib * DIM + d];
}

__global__ void init_h(const float* __restrict__ s0, float* __restrict__ h)
{
  const int idx = blockIdx.x * blockDim.x + threadIdx.x;
  if (idx < BATCH * DIM) h[idx] = s0[idx % DIM];
}

// ---------------- f32 -> bf16 converters ------------------------------
__global__ void cvt_bf16(const float* __restrict__ in, bf16_t* __restrict__ out,
                         size_t n)
{
  const size_t i = (size_t)blockIdx.x * blockDim.x + threadIdx.x;
  if (i < n) out[i] = (bf16_t)in[i];
}

__global__ void cvt_bf16_strided(const float* __restrict__ in, int inRowStride,
                                 bf16_t* __restrict__ out, int rows, int cols)
{
  const int idx = blockIdx.x * blockDim.x + threadIdx.x;
  if (idx >= rows * cols) return;
  const int r = idx / cols, c = idx % cols;
  out[(size_t)r * cols + c] = (bf16_t)in[(size_t)r * inRowStride + c];
}

// ---------------- GRU update + 2-way logit head -----------------------
__global__ __launch_bounds__(256)
void gru_head(const float* __restrict__ gi, const float* __restrict__ gh,
              float* __restrict__ h, bf16_t* __restrict__ hid_bf,
              const float* __restrict__ headW, const float* __restrict__ headb,
              float* __restrict__ logits, int t)
{
  __shared__ float s0[256], s1[256];
  const int b = blockIdx.x;
  float p0 = 0.f, p1 = 0.f;
  for (int d = threadIdx.x; d < DIM; d += 256) {
    const size_t gb = (size_t)b * TD;
    const float ir = gi[gb + d],         hr = gh[gb + d];
    const float iz = gi[gb + DIM + d],   hz = gh[gb + DIM + d];
    const float in = gi[gb + 2*DIM + d], hn = gh[gb + 2*DIM + d];
    const float r  = 1.f / (1.f + __expf(-(ir + hr)));
    const float z  = 1.f / (1.f + __expf(-(iz + hz)));
    const float c  = tanhf(in + r * hn);
    const float h2 = (1.f - z) * c + z * h[(size_t)b * DIM + d];
    h[(size_t)b * DIM + d] = h2;
    hid_bf[((size_t)b * SEQ + t) * DIM + d] = (bf16_t)h2;
    p0 += h2 * headW[d];
    p1 += h2 * headW[DIM + d];
  }
  s0[threadIdx.x] = p0; s1[threadIdx.x] = p1;
  __syncthreads();
  for (int off = 128; off > 0; off >>= 1) {
    if (threadIdx.x < off) { s0[threadIdx.x] += s0[threadIdx.x + off];
                             s1[threadIdx.x] += s1[threadIdx.x + off]; }
    __syncthreads();
  }
  if (threadIdx.x == 0) {
    logits[((size_t)b * SEQ + t) * 2 + 0] = s0[0] + headb[0];
    logits[((size_t)b * SEQ + t) * 2 + 1] = s1[0] + headb[1];
  }
}

// =====================================================================
// Host driver
// =====================================================================
static inline dim3 gemm_grid(int M, int N) { return dim3(N / 256, M / 64); }

extern "C" void kernel_launch(void* const* d_in, const int* in_sizes, int n_in,
                              void* d_out, int out_size, void* d_ws, size_t ws_size,
                              hipStream_t stream) {
  (void)in_sizes; (void)n_in; (void)out_size; (void)ws_size;
  // ---- inputs (setup_inputs order) ----
  const int*   a_seq   = (const int*)  d_in[0];
  const int*   b_seq   = (const int*)  d_in[1];
  const float* emb     = (const float*)d_in[2];
  const float* Wqkv    = (const float*)d_in[3];
  const float* bqkv    = (const float*)d_in[4];
  const float* Wo      = (const float*)d_in[5];
  const float* bo      = (const float*)d_in[6];
  const float* ln1s    = (const float*)d_in[7];
  const float* ln1b    = (const float*)d_in[8];
  const float* W1      = (const float*)d_in[9];
  const float* b1      = (const float*)d_in[10];
  const float* W2      = (const float*)d_in[11];
  const float* b2      = (const float*)d_in[12];
  const float* ln2s    = (const float*)d_in[13];
  const float* ln2b    = (const float*)d_in[14];
  const float* gWih    = (const float*)d_in[15];
  const float* gbih    = (const float*)d_in[16];
  const float* gWhh    = (const float*)d_in[17];
  const float* gbhh    = (const float*)d_in[18];
  const float* start   = (const float*)d_in[19];
  const float* outW    = (const float*)d_in[20];
  const float* outb    = (const float*)d_in[21];
  const float* headW   = (const float*)d_in[22];
  const float* headb   = (const float*)d_in[23];

  float* logits = (float*)d_out;                         // (B,S,2)
  float* soft   = (float*)d_out + (size_t)BATCH*SEQ*2;   // (B,S,D)

  // ---- workspace bump allocator ----
  char*  ws  = (char*)d_ws;
  size_t off = 0;
  auto alloc = [&](size_t bytes) -> void* {
    void* p = ws + off;
    off += (bytes + 255) & ~(size_t)255;
    return p;
  };
  bf16_t* Wqkv_bf = (bf16_t*)alloc((size_t)NL*TD*DIM*2);
  bf16_t* Wo_bf   = (bf16_t*)alloc((size_t)NL*DIM*DIM*2);
  bf16_t* W1_bf   = (bf16_t*)alloc((size_t)NL*FF*DIM*2);
  bf16_t* W2_bf   = (bf16_t*)alloc((size_t)NL*DIM*FF*2);
  bf16_t* Wih_bf  = (bf16_t*)alloc((size_t)TD*DIM*2);
  bf16_t* Whh_bf  = (bf16_t*)alloc((size_t)TD*DIM*2);
  bf16_t* outW_bf = (bf16_t*)alloc((size_t)DIM*DIM*2);
  float*  h       = (float*) alloc((size_t)BATCH*DIM*4);
  bf16_t* h_bf    = (bf16_t*)alloc((size_t)BATCH*DIM*2);
  float*  x       = (float*) alloc((size_t)BATCH*3*DIM*4);
  bf16_t* y_bf    = (bf16_t*)alloc((size_t)BATCH*3*DIM*2);
  bf16_t* act_bf  = (bf16_t*)alloc((size_t)BATCH*3*DIM*2);
  float*  qkv     = (float*) alloc((size_t)BATCH*3*TD*4);
  float*  attout  = (float*) alloc((size_t)BATCH*3*DIM*4);
  bf16_t* hrelu   = (bf16_t*)alloc((size_t)BATCH*3*FF*2);
  float*  gi      = (float*) alloc((size_t)BATCH*TD*4);
  float*  gh      = (float*) alloc((size_t)BATCH*TD*4);
  bf16_t* u_bf    = (bf16_t*)alloc((size_t)BATCH*DIM*2);
  bf16_t* hid_bf  = (bf16_t*)alloc((size_t)BATCH*SEQ*DIM*2);

  // ---- weight precision: f32 -> bf16 (per call; deterministic) ----
  auto cvt = [&](const float* src, bf16_t* dst, size_t n) {
    cvt_bf16<<<(unsigned)((n + 255) / 256), 256, 0, stream>>>(src, dst, n);
  };
  cvt(Wqkv, Wqkv_bf, (size_t)NL*TD*DIM);
  cvt(Wo,   Wo_bf,   (size_t)NL*DIM*DIM);
  cvt(W1,   W1_bf,   (size_t)NL*FF*DIM);
  cvt(W2,   W2_bf,   (size_t)NL*DIM*FF);
  cvt(gWih, Wih_bf,  (size_t)TD*DIM);
  cvt(gWhh, Whh_bf,  (size_t)TD*DIM);
  cvt(outW, outW_bf, (size_t)DIM*DIM);

  init_h<<<(BATCH*DIM + 255)/256, 256, 0, stream>>>(start, h);

  const int M3 = BATCH * 3;   // 3072 token-rows per step

  for (int t = 0; t < SEQ; ++t) {
    build_x<<<(BATCH*DIM + 255)/256, 256, 0, stream>>>(h, a_seq, b_seq, emb, x, t);

    for (int l = 0; l < NL; ++l) {
      // y = LN1(x)
      ln_to_bf16<<<M3, 256, 0, stream>>>(x, ln1s + l*DIM, ln1b + l*DIM, y_bf);
      // qkv = y @ Wqkv^T + bqkv
      gemm_bf16_wmma<<<gemm_grid(M3, TD), 256, 0, stream>>>(
          y_bf, DIM, Wqkv_bf + (size_t)l*TD*DIM, DIM,
          bqkv + l*TD, nullptr, qkv, nullptr, M3, TD, DIM, 0);
      // tiny softmax attention
      attn_small<<<(BATCH*NH + 255)/256, 256, 0, stream>>>(qkv, attout);
      // o -> bf16, x += o @ Wo^T + bo
      cvt(attout, act_bf, (size_t)M3*DIM);
      gemm_bf16_wmma<<<gemm_grid(M3, DIM), 256, 0, stream>>>(
          act_bf, DIM, Wo_bf + (size_t)l*DIM*DIM, DIM,
          bo + l*DIM, x, x, nullptr, M3, DIM, DIM, 0);
      // y = LN2(x); hrelu = relu(y @ W1^T + b1) (bf16 out)
      ln_to_bf16<<<M3, 256, 0, stream>>>(x, ln2s + l*DIM, ln2b + l*DIM, y_bf);
      gemm_bf16_wmma<<<gemm_grid(M3, FF), 256, 0, stream>>>(
          y_bf, DIM, W1_bf + (size_t)l*FF*DIM, DIM,
          b1 + l*FF, nullptr, nullptr, hrelu, M3, FF, DIM, 1);
      // x += hrelu @ W2^T + b2
      gemm_bf16_wmma<<<gemm_grid(M3, DIM), 256, 0, stream>>>(
          hrelu, FF, W2_bf + (size_t)l*DIM*FF, FF,
          b2 + l*DIM, x, x, nullptr, M3, DIM, FF, 0);
    }

    // u = x[:,0,:] ; GRU gates
    cvt_bf16_strided<<<(BATCH*DIM + 255)/256, 256, 0, stream>>>(x, 3*DIM, u_bf, BATCH, DIM);
    cvt_bf16_strided<<<(BATCH*DIM + 255)/256, 256, 0, stream>>>(h, DIM,   h_bf, BATCH, DIM);
    gemm_bf16_wmma<<<gemm_grid(BATCH, TD), 256, 0, stream>>>(
        u_bf, DIM, Wih_bf, DIM, gbih, nullptr, gi, nullptr, BATCH, TD, DIM, 0);
    gemm_bf16_wmma<<<gemm_grid(BATCH, TD), 256, 0, stream>>>(
        h_bf, DIM, Whh_bf, DIM, gbhh, nullptr, gh, nullptr, BATCH, TD, DIM, 0);
    gru_head<<<BATCH, 256, 0, stream>>>(gi, gh, h, hid_bf, headW, headb, logits, t);
  }

  // soft = hid @ outW^T + outb   (M = B*S = 65536)
  gemm_bf16_wmma<<<gemm_grid(BATCH*SEQ, DIM), 256, 0, stream>>>(
      hid_bf, DIM, outW_bf, DIM, outb, nullptr, soft, nullptr,
      BATCH*SEQ, DIM, DIM, 0);
}
